// Model_39676907887335
// MI455X (gfx1250) — matrix operand order repaired and checked by reading, not compiled, not run on hardware
//
#include <hip/hip_runtime.h>
#include <hip/hip_bf16.h>

#define BB 4
#define SS 4096
#define DD 512

typedef __bf16 bf16_t;
typedef __attribute__((ext_vector_type(16))) __bf16 v16bf;
typedef __attribute__((ext_vector_type(8)))  float  v8f;
typedef __attribute__((ext_vector_type(4)))  unsigned int v4u;
typedef __attribute__((ext_vector_type(8)))  int v8i;
typedef __attribute__((ext_vector_type(4)))  int v4i;

union Frag16 {
  v16bf v;
  uint4 q[2];
};

__device__ __forceinline__ bf16_t f2bf(float f) { return (bf16_t)f; }

// ---------------------------------------------------------------------------
// TDM 2D tile load (global -> LDS) with LDS row padding.
// D# layout per CDNA5 ISA ch.8: group0 {count, lds_addr, global_addr, type=2},
// group1 {data_size=2B, pad fields, tensor/tile dims, dim0 stride}.
// pad_interval_code: pad after 2^(code+1) DWORDs; pad_amount_code: (code+1) DWORDs.
// This toolchain exposes the 6-arg builtin (clang-23 / therock-10.0 form).
// ---------------------------------------------------------------------------
__device__ __forceinline__ void tdm_load_2d_bf16(
    unsigned int lds_addr, const void* gaddr,
    unsigned int tensor_d0, unsigned int tensor_d1,
    unsigned int tile_d0, unsigned int tile_d1,
    unsigned int stride0,
    unsigned int pad_interval_code, unsigned int pad_amount_code)
{
  const unsigned long long ga = (unsigned long long)(uintptr_t)gaddr;
  v4u g0;
  g0.x = 1u;                                           // count = 1 valid D#
  g0.y = lds_addr;                                     // LDS byte address
  g0.z = (unsigned int)ga;                             // global_addr[31:0]
  g0.w = (unsigned int)((ga >> 32) & 0x01FFFFFFu)      // global_addr[56:32]
         | (2u << 30);                                 // type = 2 ("image")
  v8i g1;
  g1[0] = (int)((1u << 16)                             // data_size = 2 bytes
              | (1u << 20)                             // pad_enable
              | (pad_interval_code << 22)
              | (pad_amount_code << 25));
  g1[1] = (int)((tensor_d0 & 0xFFFFu) << 16);          // tensor_dim0[15:0]
  g1[2] = (int)(((tensor_d0 >> 16) & 0xFFFFu)          // tensor_dim0[31:16]
              | ((tensor_d1 & 0xFFFFu) << 16));        // tensor_dim1[15:0]
  g1[3] = (int)(((tensor_d1 >> 16) & 0xFFFFu)          // tensor_dim1[31:16]
              | ((tile_d0 & 0xFFFFu) << 16));          // tile_dim0
  g1[4] = (int)(tile_d1 & 0xFFFFu);                    // tile_dim1 (tile_dim2=0)
  g1[5] = (int)stride0;                                // tensor_dim0_stride[31:0]
  g1[6] = 0;                                           // stride0 hi, stride1 lo
  g1[7] = 0;
  v4i zz4 = {0, 0, 0, 0};
  v8i zz8 = {0, 0, 0, 0, 0, 0, 0, 0};
  __builtin_amdgcn_tensor_load_to_lds(g0, g1, zz4, zz4, zz8, 0);
}

__device__ __forceinline__ unsigned int lds_addr_of(const void* p) {
  // Flat shared-aperture pointers carry the LDS byte offset in the low 32 bits.
  return (unsigned int)(uintptr_t)p;
}

// ---------------------------------------------------------------------------
// Projection GEMM: out = x @ W + b, written as bf16.
//   z==0: Qb row-major [B*S, D];  z==1: Kb row-major;  z==2: Vt [B][D][S].
// 8 waves, tile 64 rows x 128 cols; wave = 16 rows x 64 cols.
// ---------------------------------------------------------------------------
__global__ __launch_bounds__(256) void proj_kernel(
    const float* __restrict__ q_in, const float* __restrict__ k_in,
    const float* __restrict__ v_in,
    const float* __restrict__ Wq, const float* __restrict__ bq,
    const float* __restrict__ Wk, const float* __restrict__ bk,
    const float* __restrict__ Wv, const float* __restrict__ bvv,
    bf16_t* __restrict__ Qb, bf16_t* __restrict__ Kb, bf16_t* __restrict__ Vt)
{
  const int z = blockIdx.z;
  const float* __restrict__ x    = (z == 0) ? q_in : (z == 1) ? k_in : v_in;
  const float* __restrict__ W    = (z == 0) ? Wq   : (z == 1) ? Wk   : Wv;
  const float* __restrict__ bias = (z == 0) ? bq   : (z == 1) ? bk   : bvv;

  const int r0 = blockIdx.y * 64;
  const int c0 = blockIdx.x * 128;

  __shared__ bf16_t As[64 * 40];    // 64 rows x 32 k (stride 40 kills conflicts)
  __shared__ bf16_t Bt[128 * 40];   // Bt[n][k] = W[k][n]

  const int t    = threadIdx.x;
  const int lane = t & 31;
  const int w    = t >> 5;
  const int rg   = w & 3;
  const int cgI  = w >> 2;

  const int m    = lane & 15;
  const int half = lane >> 4;
  const int kb2  = half * 8;

  v8f z8 = {0.f,0.f,0.f,0.f,0.f,0.f,0.f,0.f};
  v8f acc[4];
  #pragma unroll
  for (int i = 0; i < 4; i++) acc[i] = z8;

  for (int k0 = 0; k0 < DD; k0 += 32) {
    {
      const int row = t >> 2;
      const int kk  = (t & 3) * 8;
      const float* src = &x[(size_t)(r0 + row) * DD + k0 + kk];
      float4 f0 = *(const float4*)(src);
      float4 f1 = *(const float4*)(src + 4);
      bf16_t* d = &As[row * 40 + kk];
      d[0]=f2bf(f0.x); d[1]=f2bf(f0.y); d[2]=f2bf(f0.z); d[3]=f2bf(f0.w);
      d[4]=f2bf(f1.x); d[5]=f2bf(f1.y); d[6]=f2bf(f1.z); d[7]=f2bf(f1.w);
    }
    {
      const int kk = t >> 3;
      const int n0 = (t & 7) * 16;
      const float* src = &W[(size_t)(k0 + kk) * DD + c0 + n0];
      #pragma unroll
      for (int i = 0; i < 16; i += 4) {
        float4 f = *(const float4*)(src + i);
        Bt[(n0 + i + 0) * 40 + kk] = f2bf(f.x);
        Bt[(n0 + i + 1) * 40 + kk] = f2bf(f.y);
        Bt[(n0 + i + 2) * 40 + kk] = f2bf(f.z);
        Bt[(n0 + i + 3) * 40 + kk] = f2bf(f.w);
      }
    }
    __syncthreads();

    // Hoist A + all 4 B fragments, then burst the wmmas (batched dscnt waits).
    Frag16 a;
    const bf16_t* ap = &As[(rg * 16 + m) * 40 + kb2];
    a.q[0] = *(const uint4*)(ap);
    a.q[1] = *(const uint4*)(ap + 16);

    Frag16 b[4];
    #pragma unroll
    for (int nt = 0; nt < 4; nt++) {
      const bf16_t* bp = &Bt[(cgI * 64 + nt * 16 + m) * 40 + half * 8];
      b[nt].q[0] = *(const uint4*)(bp);
      b[nt].q[1] = *(const uint4*)(bp + 16);
    }
    #pragma unroll
    for (int nt = 0; nt < 4; nt++) {
      acc[nt] = __builtin_amdgcn_wmma_f32_16x16x32_bf16(
          false, a.v, false, b[nt].v, (short)0, acc[nt], false, false);
    }
    __syncthreads();
  }

  #pragma unroll
  for (int nt = 0; nt < 4; nt++) {
    const int col = c0 + cgI * 64 + nt * 16 + m;
    const float bval = bias[col];
    #pragma unroll
    for (int j = 0; j < 8; j++) {
      const int row  = r0 + rg * 16 + j + half * 8;
      const float vr = acc[nt][j] + bval;
      if (z == 0) {
        Qb[(size_t)row * DD + col] = f2bf(vr);
      } else if (z == 1) {
        Kb[(size_t)row * DD + col] = f2bf(vr);
      } else {
        const int b2 = row >> 12;
        const int s  = row & 4095;
        Vt[((size_t)(b2 * DD + col)) * SS + s] = f2bf(vr);
      }
    }
  }
}

// ---------------------------------------------------------------------------
// Flash attention, 32 query rows x D=512 per workgroup (8 waves).
// Q tile + K blocks staged by the Tensor Data Mover with D# LDS padding
// (row stride 520 bf16) and TENSORcnt tracking; the K load for block kb+1 is
// issued right after all Ks reads of block kb finish, overlapping DMA with
// softmax + PV. V fragments stream from L2-resident global bf16.
// ---------------------------------------------------------------------------
__global__ __launch_bounds__(256) void attn_kernel(
    const bf16_t* __restrict__ Qb, const bf16_t* __restrict__ Kb,
    const bf16_t* __restrict__ Vt, const float* __restrict__ inv_scale,
    float* __restrict__ out)
{
  const int bb = blockIdx.y;
  const int q0 = blockIdx.x * 32;

  __shared__ bf16_t Qs[32 * 520];     // TDM-padded stride 520
  __shared__ bf16_t Ks[64 * 520];     // TDM-padded stride 520
  __shared__ bf16_t Ps[32 * 72];
  __shared__ float  m_prev[32], m_cur[32], alpha_s[32], l_sum[32];
  __shared__ float  rowmaxP[32][4];
  __shared__ float  rowsumP[32][4];

  const int t    = threadIdx.x;
  const int lane = t & 31;
  const int w    = t >> 5;
  const int rg   = w >> 2;
  const int cg   = w & 3;

  const int m    = lane & 15;
  const int half = lane >> 4;
  const int kb2  = half * 8;

  const float rscale = 1.0f / inv_scale[0];

  // Prologue: TDM-stage the Q tile and the first K block (wave 0 issues).
  if (w == 0) {
    tdm_load_2d_bf16(lds_addr_of(Qs), &Qb[((size_t)(bb * SS + q0)) * DD],
                     /*tensor*/ DD, SS, /*tile*/ DD, 32, /*stride0*/ DD,
                     /*pad: 256 DW interval*/ 7, /*4 DW amount*/ 3);
    tdm_load_2d_bf16(lds_addr_of(Ks), &Kb[((size_t)(bb * SS)) * DD],
                     DD, SS, DD, 64, DD, 7, 3);
  }
  if (t < 32) { m_prev[t] = -__builtin_inff(); l_sum[t] = 0.f; }

  v8f z8 = {0.f,0.f,0.f,0.f,0.f,0.f,0.f,0.f};
  v8f acc[8];
  #pragma unroll
  for (int i = 0; i < 8; i++) acc[i] = z8;

  for (int kb = 0; kb < SS / 64; kb++) {
    if (w == 0) __builtin_amdgcn_s_wait_tensorcnt(0);
    __syncthreads();                       // Ks(kb) + (first iter) Qs ready

    // Prefetch next V block toward L2.
    if (kb + 1 < SS / 64) {
      const int dpre = (t * 2) & (DD - 1);
      __builtin_prefetch(&Vt[((size_t)(bb * DD + dpre)) * SS + (kb + 1) * 64], 0, 1);
    }

    // ---- S = (Q K^T) * rscale : one 16x16 tile per wave, K from LDS ----
    v8f sacc = z8;
    const bf16_t* ks_row = &Ks[(cg * 16 + m) * 520 + half * 8];
    #pragma unroll 4
    for (int d0 = 0; d0 < DD; d0 += 32) {
      Frag16 a, b;
      const bf16_t* ap = &Qs[(rg * 16 + m) * 520 + d0 + kb2];
      a.q[0] = *(const uint4*)(ap);
      a.q[1] = *(const uint4*)(ap + 16);
      const bf16_t* bp = ks_row + d0;
      b.q[0] = *(const uint4*)(bp);
      b.q[1] = *(const uint4*)(bp + 16);
      sacc = __builtin_amdgcn_wmma_f32_16x16x32_bf16(
          false, a.v, false, b.v, (short)0, sacc, false, false);
    }

    // Row maxima: reduce across the 16 lanes of each half.
    float sv[8];
    #pragma unroll
    for (int j = 0; j < 8; j++) {
      sv[j] = sacc[j] * rscale;
      float v = sv[j];
      #pragma unroll
      for (int off = 1; off < 16; off <<= 1) v = fmaxf(v, __shfl_xor(v, off, 32));
      if (m == 0) rowmaxP[rg * 16 + j + half * 8][cg] = v;
    }
    __syncthreads();                       // B1: all Ks reads done

    // Overlap: kick the DMA for the next K block while softmax + PV run.
    if (w == 0 && kb + 1 < SS / 64) {
      tdm_load_2d_bf16(lds_addr_of(Ks),
                       &Kb[((size_t)(bb * SS + (kb + 1) * 64)) * DD],
                       DD, SS, DD, 64, DD, 7, 3);
    }
    if (t < 32) {
      float mn = fmaxf(fmaxf(rowmaxP[t][0], rowmaxP[t][1]),
                       fmaxf(rowmaxP[t][2], rowmaxP[t][3]));
      mn = fmaxf(m_prev[t], mn);
      alpha_s[t] = __expf(m_prev[t] - mn);
      m_cur[t]   = mn;
      m_prev[t]  = mn;
    }
    __syncthreads();                       // B2

    // p = exp(s - m_new); P -> LDS (bf16); partial sums; rescale O.
    #pragma unroll
    for (int j = 0; j < 8; j++) {
      const int row = rg * 16 + j + half * 8;
      const float p = __expf(sv[j] - m_cur[row]);
      Ps[row * 72 + cg * 16 + m] = f2bf(p);
      float v = p;
      #pragma unroll
      for (int off = 1; off < 16; off <<= 1) v += __shfl_xor(v, off, 32);
      if (m == 0) rowsumP[row][cg] = v;
      const float al = alpha_s[row];
      #pragma unroll
      for (int nt = 0; nt < 8; nt++) acc[nt][j] *= al;
    }
    __syncthreads();                       // B3

    if (t < 32) {
      l_sum[t] = l_sum[t] * alpha_s[t] +
                 rowsumP[t][0] + rowsumP[t][1] + rowsumP[t][2] + rowsumP[t][3];
    }

    // ---- O += P @ V_block : hoist 4 B-frags at a time, then wmma burst ----
    #pragma unroll
    for (int ks = 0; ks < 2; ks++) {
      Frag16 a;
      const bf16_t* ap = &Ps[(rg * 16 + m) * 72 + ks * 32 + kb2];
      a.q[0] = *(const uint4*)(ap);
      a.q[1] = *(const uint4*)(ap + 16);
      const int keyb = kb * 64 + ks * 32 + half * 8;
      #pragma unroll
      for (int g = 0; g < 2; g++) {
        Frag16 b[4];
        #pragma unroll
        for (int i = 0; i < 4; i++) {
          const int d = cg * 128 + (g * 4 + i) * 16 + m;
          const bf16_t* bp = &Vt[((size_t)(bb * DD + d)) * SS + keyb];
          b[i].q[0] = *(const uint4*)(bp);
          b[i].q[1] = *(const uint4*)(bp + 16);
        }
        #pragma unroll
        for (int i = 0; i < 4; i++) {
          acc[g * 4 + i] = __builtin_amdgcn_wmma_f32_16x16x32_bf16(
              false, a.v, false, b[i].v, (short)0, acc[g * 4 + i], false, false);
        }
      }
    }
  }
  __syncthreads();

  // Epilogue: O / l, f32 store.
  #pragma unroll
  for (int nt = 0; nt < 8; nt++) {
    const int d = cg * 128 + nt * 16 + m;
    #pragma unroll
    for (int j = 0; j < 8; j++) {
      const int row = rg * 16 + j + half * 8;
      out[((size_t)(bb * SS + q0 + row)) * DD + d] = acc[nt][j] / l_sum[row];
    }
  }
}

extern "C" void kernel_launch(void* const* d_in, const int* in_sizes, int n_in,
                              void* d_out, int out_size, void* d_ws, size_t ws_size,
                              hipStream_t stream) {
  const float* query     = (const float*)d_in[0];
  const float* key       = (const float*)d_in[1];
  const float* value     = (const float*)d_in[2];
  const float* inv_scale = (const float*)d_in[3];
  const float* Wq        = (const float*)d_in[4];
  const float* bq        = (const float*)d_in[5];
  const float* Wk        = (const float*)d_in[6];
  const float* bk        = (const float*)d_in[7];
  const float* Wv        = (const float*)d_in[8];
  const float* bv        = (const float*)d_in[9];
  float* out = (float*)d_out;

  bf16_t* Qb = (bf16_t*)d_ws;                 // [B*S, D] bf16
  bf16_t* Kb = Qb + (size_t)BB * SS * DD;     // [B*S, D] bf16
  bf16_t* Vt = Kb + (size_t)BB * SS * DD;     // [B, D, S] bf16

  dim3 pgrid(DD / 128, (BB * SS) / 64, 3);
  proj_kernel<<<pgrid, dim3(256), 0, stream>>>(query, key, value,
                                               Wq, bq, Wk, bk, Wv, bv,
                                               Qb, Kb, Vt);

  dim3 agrid(SS / 32, BB);
  attn_kernel<<<agrid, dim3(256), 0, stream>>>(Qb, Kb, Vt, inv_scale, out);
}
